// BigramTransformer_16346645528839
// MI455X (gfx1250) — matrix operand order repaired and checked by reading, not compiled, not run on hardware
//
#include <hip/hip_runtime.h>
#include <hip/hip_bf16.h>

// ---------------------------------------------------------------------------
// GPT-style transformer forward on gfx1250 (MI455X), built around
// V_WMMA_F32_16X16X32_F16.  All GEMMs (QKV, attention scores, attn*V,
// out-proj, FFN1, FFN2, LM head) run on the matrix pipes.
//
// Fragment layouts (CDNA5 ISA 7.12.2, wave32):
//   A 16x32 f16 : lane l, m = l&15, hi = l>>4.
//                 v16h elems 0..7  = A[m][8*hi + 0..7]
//                 v16h elems 8..15 = A[m][16 + 8*hi + 0..7]     -> 2x b128
//   B 32x16 f16 : lane l, n = l&15, hi = l>>4.
//                 v16h elems 0..7 = B[8*hi + 0..7][n], 8..15 = B[16 + 8*hi + 0..7][n]
//                 i.e. contiguous in K if B is stored transposed (BT[n][k])
//                                                               -> 2x b128
//   C/D 16x16 f32: lane l, n = l&15; acc[r] is row m = 8*(l>>4) + r.
//
// GEMM is register-blocked: each wave owns a 4(M) x 2(N) grid of 16x16 tiles,
// so each K-step loads 4 A-frags + 2 B-frags and issues 8 WMMAs (2.7x the
// operand reuse of a 1-tile wave).  8 waves/WG cover a 128x128 output tile.
// ---------------------------------------------------------------------------

typedef _Float16 half_t;
typedef __attribute__((ext_vector_type(16))) _Float16 v16h;
typedef __attribute__((ext_vector_type(8)))  _Float16 v8h;
typedef __attribute__((ext_vector_type(8)))  float    v8f;

union H16 { v16h v; v8h h[2]; };

#define NB    2048
#define TT    64
#define DD    128
#define HH    4
#define HDIM  32
#define LL    6
#define VV    256
#define DFF_  512
#define NTOK  (NB * TT)          // 131072 rows

enum { MODE_F16 = 0, MODE_RELU_F16 = 1, MODE_RES_F32 = 2, MODE_F32 = 3 };

__device__ __forceinline__ v16h cat8(v8h a, v8h b) {
    H16 u; u.h[0] = a; u.h[1] = b; return u.v;
}

// A-fragment loader: arow = &A[mtile*16 + (lane&15)][0], row-major, ld >= k0+32
__device__ __forceinline__ v16h load_afrag(const half_t* arow, int k0, int hi) {
    v8h a0 = *(const v8h*)(arow + k0 + 8 * hi);
    v8h a1 = *(const v8h*)(arow + k0 + 16 + 8 * hi);
    return cat8(a0, a1);
}

// B-fragment loader: brow = &BT[ntile*16 + (lane&15)][0], BT row-major in K
__device__ __forceinline__ v16h load_bfrag(const half_t* brow, int k0, int hi) {
    v8h b0 = *(const v8h*)(brow + k0 + 8 * hi);
    v8h b1 = *(const v8h*)(brow + k0 + 16 + 8 * hi);
    return cat8(b0, b1);
}

__device__ __forceinline__ v8f wmma_acc(v16h a, v16h b, v8f acc) {
    return __builtin_amdgcn_wmma_f32_16x16x32_f16(false, a, false, b, (short)0,
                                                  acc, false, false);
}

// One-tile step (used by the attention kernel)
__device__ __forceinline__ v8f wmma_step(const half_t* arow, const half_t* brow,
                                         int k0, int hi, v8f acc) {
    return wmma_acc(load_afrag(arow, k0, hi), load_bfrag(brow, k0, hi), acc);
}

// ---------------------------------------------------------------------------
// x[n][d] = tok_emb[idx[n]][d] + pos_emb[n % T][d]
// ---------------------------------------------------------------------------
__global__ void embed_kernel(const int* __restrict__ idx, const float* __restrict__ tok,
                             const float* __restrict__ pos, float* __restrict__ x) {
    int i = blockIdx.x * 256 + threadIdx.x;
    int n = i >> 7, d = i & 127;
    x[(size_t)n * DD + d] = tok[(size_t)idx[n] * DD + d] + pos[(n & (TT - 1)) * DD + d];
}

// ---------------------------------------------------------------------------
// Batched f32 -> f16 transpose: src[b][k][n] -> dst[b][n][k]
// (stages every weight matrix in the BT layout the B-fragment loader wants)
// ---------------------------------------------------------------------------
__global__ void convert_transpose(const float* __restrict__ src, half_t* __restrict__ dst,
                                  int K, int Ncol, int total) {
    int i = blockIdx.x * 256 + threadIdx.x;
    if (i >= total) return;
    int k = i % K;
    int bn = i / K;
    int n = bn % Ncol;
    int b = bn / Ncol;
    dst[i] = (half_t)src[((size_t)b * K + k) * Ncol + n];
}

// ---------------------------------------------------------------------------
// LayerNorm over D=128, one wave per row, f16 output. Biased variance.
// ---------------------------------------------------------------------------
__global__ void __launch_bounds__(256)
ln_kernel(const float* __restrict__ x, const float* __restrict__ g,
          const float* __restrict__ bta, half_t* __restrict__ out) {
    int wave = threadIdx.x >> 5, l = threadIdx.x & 31;
    size_t row = (size_t)blockIdx.x * 8 + wave;
    float4 xv = ((const float4*)(x + row * DD))[l];
    float s = xv.x + xv.y + xv.z + xv.w;
#pragma unroll
    for (int o = 16; o >= 1; o >>= 1) s += __shfl_xor(s, o, 32);
    float mean = s * (1.0f / DD);
    float dx0 = xv.x - mean, dx1 = xv.y - mean, dx2 = xv.z - mean, dx3 = xv.w - mean;
    float vs = dx0 * dx0 + dx1 * dx1 + dx2 * dx2 + dx3 * dx3;
#pragma unroll
    for (int o = 16; o >= 1; o >>= 1) vs += __shfl_xor(vs, o, 32);
    float inv = rsqrtf(vs * (1.0f / DD) + 1e-5f);
    float4 gv = ((const float4*)g)[l];
    float4 bv = ((const float4*)bta)[l];
    half_t* orow = out + row * DD + 4 * l;
    orow[0] = (half_t)(dx0 * inv * gv.x + bv.x);
    orow[1] = (half_t)(dx1 * inv * gv.y + bv.y);
    orow[2] = (half_t)(dx2 * inv * gv.z + bv.z);
    orow[3] = (half_t)(dx3 * inv * gv.w + bv.w);
}

// ---------------------------------------------------------------------------
// Register-blocked WMMA GEMM: out[M][Ncols] = A[M][K] @ BT[Ncols][K]^T (+bias)
// Block = 256 threads = 8 waves laid out 2(M) x 4(N); each wave computes a
// 4x2 grid of 16x16 tiles -> workgroup tile = 128 rows x 128 cols.
// grid = (M/128, Ncols/128).
// ---------------------------------------------------------------------------
template <int K, int MODE>
__global__ void __launch_bounds__(256)
gemm_rb(const half_t* __restrict__ A, const half_t* __restrict__ BT,
        const float* __restrict__ bias, half_t* __restrict__ out16,
        float* __restrict__ outf, int Ncols) {
    int wave = threadIdx.x >> 5, l = threadIdx.x & 31;
    int nl = l & 15, hi = l >> 4;
    int mw = wave & 1;        // 2 waves along M
    int nw = wave >> 1;       // 4 waves along N
    int tm0 = blockIdx.x * 8 + mw * 4;   // first of 4 M-tiles
    int tn0 = blockIdx.y * 8 + nw * 2;   // first of 2 N-tiles

    const half_t* ar[4];
    const half_t* br[2];
#pragma unroll
    for (int i = 0; i < 4; ++i) ar[i] = A + ((size_t)(tm0 + i) * 16 + nl) * K;
#pragma unroll
    for (int j = 0; j < 2; ++j) br[j] = BT + ((size_t)(tn0 + j) * 16 + nl) * K;

    v8f acc[4][2];
#pragma unroll
    for (int i = 0; i < 4; ++i)
#pragma unroll
        for (int j = 0; j < 2; ++j) acc[i][j] = (v8f){};

    for (int ks = 0; ks < K / 32; ++ks) {
        int k0 = ks * 32;
        v16h af[4], bf[2];
#pragma unroll
        for (int i = 0; i < 4; ++i) af[i] = load_afrag(ar[i], k0, hi);
#pragma unroll
        for (int j = 0; j < 2; ++j) bf[j] = load_bfrag(br[j], k0, hi);
#pragma unroll
        for (int i = 0; i < 4; ++i)
#pragma unroll
            for (int j = 0; j < 2; ++j) acc[i][j] = wmma_acc(af[i], bf[j], acc[i][j]);
    }

#pragma unroll
    for (int j = 0; j < 2; ++j) {
        int n = (tn0 + j) * 16 + nl;
        float bv = bias ? bias[n] : 0.0f;
#pragma unroll
        for (int i = 0; i < 4; ++i) {
#pragma unroll
            for (int r = 0; r < 8; ++r) {
                size_t m = (size_t)(tm0 + i) * 16 + 8 * hi + r;
                float val = acc[i][j][r] + bv;
                if constexpr (MODE == MODE_RELU_F16) val = fmaxf(val, 0.0f);
                if constexpr (MODE == MODE_RES_F32) {
                    outf[m * Ncols + n] += val;
                } else if constexpr (MODE == MODE_F32) {
                    outf[m * Ncols + n] = val;
                } else {
                    out16[m * Ncols + n] = (half_t)val;
                }
            }
        }
    }
}

// ---------------------------------------------------------------------------
// Fused causal attention for one (batch, head): S = QK^T/sqrt(32), causal
// softmax, O = P V.  q/k/v/o are [NTOK][128] f16, head slice cols h*32..+31.
// ---------------------------------------------------------------------------
__global__ void __launch_bounds__(256)
attn_kernel(const half_t* __restrict__ q, const half_t* __restrict__ k,
            const half_t* __restrict__ v, half_t* __restrict__ o) {
    __shared__ __align__(16) float  sS[TT * TT];     // 16 KB scores
    __shared__ __align__(16) half_t sW[TT * TT];     // 8 KB probs (A-frags)
    __shared__ __align__(16) half_t sVT[HDIM * TT];  // 4 KB V^T (B-frags)

    int tid = threadIdx.x, wave = tid >> 5, l = tid & 31;
    int nl = l & 15, hi = l >> 4;
    int b = blockIdx.x >> 2, h = blockIdx.x & 3;
    size_t base = ((size_t)b * TT) * DD + (size_t)h * HDIM;

    // V^T into LDS: sVT[n][t] = v[t][n]
    for (int i = tid; i < TT * HDIM; i += 256) {
        int t = i >> 5, n = i & 31;
        sVT[n * TT + t] = v[base + (size_t)t * DD + n];
    }

    // S tiles (4x4 grid of 16x16); K^T columns == K rows -> BT layout for free
    for (int tile = wave; tile < 16; tile += 8) {
        int ti = tile >> 2, tj = tile & 3;
        const half_t* arow = q + base + (size_t)(ti * 16 + nl) * DD;
        const half_t* brow = k + base + (size_t)(tj * 16 + nl) * DD;
        v8f acc = {};
        acc = wmma_step(arow, brow, 0, hi, acc);   // K = HD = 32, one WMMA
#pragma unroll
        for (int r = 0; r < 8; ++r)
            sS[(ti * 16 + 8 * hi + r) * TT + tj * 16 + nl] = acc[r] * 0.17677669529663689f;
    }
    __syncthreads();

    // Causal softmax, one wave per 8 rows, 2 columns per lane
    for (int rr = 0; rr < 8; ++rr) {
        int row = wave * 8 + rr;
        float x0 = sS[row * TT + l];
        float x1 = sS[row * TT + 32 + l];
        if (l > row) x0 = -1e30f;
        if (32 + l > row) x1 = -1e30f;
        float mx = fmaxf(x0, x1);
#pragma unroll
        for (int off = 16; off >= 1; off >>= 1) mx = fmaxf(mx, __shfl_xor(mx, off, 32));
        float e0 = __expf(x0 - mx), e1 = __expf(x1 - mx);
        float sm = e0 + e1;
#pragma unroll
        for (int off = 16; off >= 1; off >>= 1) sm += __shfl_xor(sm, off, 32);
        float inv = 1.0f / sm;
        sW[row * TT + l] = (half_t)(e0 * inv);
        sW[row * TT + 32 + l] = (half_t)(e1 * inv);
    }
    __syncthreads();

    // O tiles: 4 M-tiles x 2 N-tiles = 8 tiles, one per wave; K=64 -> 2 WMMAs
    int ti = wave >> 1, nt = wave & 1;
    const half_t* arow = &sW[(ti * 16 + nl) * TT];
    const half_t* brow = &sVT[(nt * 16 + nl) * TT];
    v8f acc = {};
    acc = wmma_step(arow, brow, 0, hi, acc);
    acc = wmma_step(arow, brow, 32, hi, acc);
#pragma unroll
    for (int r = 0; r < 8; ++r)
        o[base + (size_t)(ti * 16 + 8 * hi + r) * DD + nt * 16 + nl] = (half_t)acc[r];
}

// ---------------------------------------------------------------------------
extern "C" void kernel_launch(void* const* d_in, const int* in_sizes, int n_in,
                              void* d_out, int out_size, void* d_ws, size_t ws_size,
                              hipStream_t stream) {
    (void)in_sizes; (void)n_in; (void)out_size; (void)ws_size;

    const int*   idx  = (const int*)  d_in[0];
    const float* tok  = (const float*)d_in[1];
    const float* pos  = (const float*)d_in[2];
    const float* Wq   = (const float*)d_in[3];
    const float* Wk   = (const float*)d_in[4];
    const float* Wv   = (const float*)d_in[5];
    const float* Wo   = (const float*)d_in[6];
    const float* bo   = (const float*)d_in[7];
    const float* ln1g = (const float*)d_in[8];
    const float* ln1b = (const float*)d_in[9];
    const float* W1   = (const float*)d_in[10];
    const float* b1   = (const float*)d_in[11];
    const float* W2   = (const float*)d_in[12];
    const float* b2   = (const float*)d_in[13];
    const float* ln2g = (const float*)d_in[14];
    const float* ln2b = (const float*)d_in[15];
    const float* lnfg = (const float*)d_in[16];
    const float* lnfb = (const float*)d_in[17];
    const float* Wh   = (const float*)d_in[18];
    const float* bh   = (const float*)d_in[19];
    float* logits = (float*)d_out;

    // Workspace carve-up (~237 MB total)
    char* ws = (char*)d_ws;
    float*  x    = (float*)ws;  ws += (size_t)NTOK * DD * sizeof(float);   // 67 MB
    half_t* hbuf = (half_t*)ws; ws += (size_t)NTOK * DD * sizeof(half_t);  // 33 MB
    half_t* qbuf = (half_t*)ws; ws += (size_t)NTOK * DD * sizeof(half_t);
    half_t* kbuf = (half_t*)ws; ws += (size_t)NTOK * DD * sizeof(half_t);
    half_t* vbuf = (half_t*)ws; ws += (size_t)NTOK * DD * sizeof(half_t);
    half_t* obuf = (half_t*)ws; ws += (size_t)NTOK * DD * sizeof(half_t);
    half_t* ubuf = qbuf;  // FFN hidden [NTOK][512] aliases q/k/v/o (dead by then)
    half_t* WqT = (half_t*)ws; ws += (size_t)LL * DD * DD * sizeof(half_t);
    half_t* WkT = (half_t*)ws; ws += (size_t)LL * DD * DD * sizeof(half_t);
    half_t* WvT = (half_t*)ws; ws += (size_t)LL * DD * DD * sizeof(half_t);
    half_t* WoT = (half_t*)ws; ws += (size_t)LL * DD * DD * sizeof(half_t);
    half_t* W1T = (half_t*)ws; ws += (size_t)LL * DD * DFF_ * sizeof(half_t);
    half_t* W2T = (half_t*)ws; ws += (size_t)LL * DFF_ * DD * sizeof(half_t);
    half_t* WhT = (half_t*)ws; ws += (size_t)DD * VV * sizeof(half_t);

    dim3 blk(256);
    const int MB128 = NTOK / 128;  // 1024 workgroup M-tiles

    // Embedding
    embed_kernel<<<(NTOK * DD) / 256, blk, 0, stream>>>(idx, tok, pos, x);

    // Stage weights as f16 transposed (BT layout).
    // Wq/Wk/Wv [L,H,D,HD]: batch=L*H, K=D, N=HD -> dst [l][h*32+e][d] = combined WqT
    convert_transpose<<<(LL * HH * DD * HDIM) / 256, blk, 0, stream>>>(Wq, WqT, DD, HDIM, LL * HH * DD * HDIM);
    convert_transpose<<<(LL * HH * DD * HDIM) / 256, blk, 0, stream>>>(Wk, WkT, DD, HDIM, LL * HH * DD * HDIM);
    convert_transpose<<<(LL * HH * DD * HDIM) / 256, blk, 0, stream>>>(Wv, WvT, DD, HDIM, LL * HH * DD * HDIM);
    convert_transpose<<<(LL * DD * DD) / 256, blk, 0, stream>>>(Wo, WoT, DD, DD, LL * DD * DD);
    convert_transpose<<<(LL * DD * DFF_) / 256, blk, 0, stream>>>(W1, W1T, DD, DFF_, LL * DD * DFF_);
    convert_transpose<<<(LL * DFF_ * DD) / 256, blk, 0, stream>>>(W2, W2T, DFF_, DD, LL * DFF_ * DD);
    convert_transpose<<<(DD * VV) / 256, blk, 0, stream>>>(Wh, WhT, DD, VV, DD * VV);

    for (int l = 0; l < LL; ++l) {
        const size_t wdd = (size_t)l * DD * DD;
        const size_t wdf = (size_t)l * DD * DFF_;

        // h = LN1(x)
        ln_kernel<<<NTOK / 8, blk, 0, stream>>>(x, ln1g + l * DD, ln1b + l * DD, hbuf);
        // q/k/v = h @ W{q,k,v}   (K=128, N=128)
        gemm_rb<DD, MODE_F16><<<dim3(MB128, 1), blk, 0, stream>>>(hbuf, WqT + wdd, nullptr, qbuf, nullptr, DD);
        gemm_rb<DD, MODE_F16><<<dim3(MB128, 1), blk, 0, stream>>>(hbuf, WkT + wdd, nullptr, kbuf, nullptr, DD);
        gemm_rb<DD, MODE_F16><<<dim3(MB128, 1), blk, 0, stream>>>(hbuf, WvT + wdd, nullptr, vbuf, nullptr, DD);
        // o = causal_softmax(q k^T / sqrt(HD)) v
        attn_kernel<<<NB * HH, blk, 0, stream>>>(qbuf, kbuf, vbuf, obuf);
        // x += o @ Wo + bo
        gemm_rb<DD, MODE_RES_F32><<<dim3(MB128, 1), blk, 0, stream>>>(obuf, WoT + wdd, bo + l * DD, nullptr, x, DD);
        // h = LN2(x)
        ln_kernel<<<NTOK / 8, blk, 0, stream>>>(x, ln2g + l * DD, ln2b + l * DD, hbuf);
        // u = relu(h @ W1 + b1)   (K=128, N=512)
        gemm_rb<DD, MODE_RELU_F16><<<dim3(MB128, 4), blk, 0, stream>>>(hbuf, W1T + wdf, b1 + l * DFF_, ubuf, nullptr, DFF_);
        // x += u @ W2 + b2        (K=512, N=128)
        gemm_rb<DFF_, MODE_RES_F32><<<dim3(MB128, 1), blk, 0, stream>>>(ubuf, W2T + wdf, b2 + l * DD, nullptr, x, DD);
    }

    // Final LN + LM head: logits = LN_f(x) @ Wh + bh  (K=128, N=256)
    ln_kernel<<<NTOK / 8, blk, 0, stream>>>(x, lnfg, lnfb, hbuf);
    gemm_rb<DD, MODE_F32><<<dim3(MB128, 2), blk, 0, stream>>>(hbuf, WhT, bh, nullptr, logits, VV);
}
